// MultiHeadAttention_45329084842324
// MI455X (gfx1250) — matrix operand order repaired and checked by reading, not compiled, not run on hardware
//
#include <hip/hip_runtime.h>
#include <cstdint>
#include <cstddef>

#define D_MODEL 1024
#define SEQ     2048
#define NHEADS  16
#define DKH     64
#define NB      2
#define ROWS    (NB*SEQ)       /* 4096 */
#define NEGV    (-1e9f)

typedef __attribute__((ext_vector_type(16))) __bf16       v16bf;
typedef __attribute__((ext_vector_type(8)))  float        v8f;
typedef __attribute__((ext_vector_type(4)))  unsigned int u32x4;
typedef __attribute__((ext_vector_type(2)))  unsigned int u32x2;
typedef __attribute__((ext_vector_type(4)))  float        f32x4;

union FragBF { u32x4 q[2]; v16bf v; };

__device__ __forceinline__ unsigned short bf16b(float f) {
    unsigned u = __builtin_bit_cast(unsigned, f);
    u += 0x7FFFu + ((u >> 16) & 1u);            // round-to-nearest-even
    return (unsigned short)(u >> 16);
}
__device__ __forceinline__ unsigned pack2(float a, float b) {
    return (unsigned)bf16b(a) | ((unsigned)bf16b(b) << 16);
}
__device__ __forceinline__ void zero8(v8f& a) {
#pragma unroll
    for (int e = 0; e < 8; ++e) a[e] = 0.0f;
}

// Async global->LDS copy (GVS mode: SGPR 64-bit base + per-lane 32-bit byte off).
// LDS dest address = low 32 bits of the generic pointer to the __shared__ object
// (ISA 10.2: LDS aperture addresses are truncated to [31:0] for the LDS index).
__device__ __forceinline__ void async_ld_b64(unsigned lds_addr, const void* gbase,
                                             unsigned gbyte_off) {
    asm volatile("global_load_async_to_lds_b64 %0, %1, %2"
                 :: "v"(lds_addr), "v"(gbyte_off), "s"(gbase)
                 : "memory");
}
__device__ __forceinline__ void wait_async0() {
    asm volatile("s_wait_asynccnt 0x0" ::: "memory");
}

// ---------------------------------------------------------------------------
// Stage 1: C[M=4096,N=1024] = X @ W + bias, output bf16 head-split [B,H,S,DK]
// ---------------------------------------------------------------------------
__global__ __launch_bounds__(256)
void qkv_proj_kernel(const float* __restrict__ X, const float* __restrict__ W,
                     const float* __restrict__ bias, unsigned short* __restrict__ Out)
{
    __shared__ unsigned short As[128 * 40];   // [m][k] bf16, stride 40 (80B, 16B aligned)
    __shared__ unsigned short Bs[128 * 40];   // [n][k] bf16
    const int m0 = blockIdx.x * 128, n0 = blockIdx.y * 128;
    const int t = threadIdx.x, lane = t & 31, w = t >> 5;
    const int g = lane >> 4, ln = lane & 15;
    const int wm = w >> 1, wn = w & 1;        // 4x2 waves -> 32x64 per wave

    v8f acc[2][4];
#pragma unroll
    for (int i = 0; i < 2; ++i)
#pragma unroll
        for (int j = 0; j < 4; ++j) zero8(acc[i][j]);

    for (int k0 = 0; k0 < D_MODEL; k0 += 32) {
        // stage A tile: 128x32 fp32 -> bf16
#pragma unroll
        for (int i = 0; i < 4; ++i) {
            int e = t + i * 256;              // 1024 float4 slots
            int r = e >> 3, c = (e & 7) << 2;
            f32x4 v4 = *(const f32x4*)(X + (size_t)(m0 + r) * D_MODEL + k0 + c);
            u32x2 p; p.x = pack2(v4[0], v4[1]); p.y = pack2(v4[2], v4[3]);
            *(u32x2*)&As[r * 40 + c] = p;
        }
        // stage B tile: W[k][n] -> Bs[n][k]
#pragma unroll
        for (int i = 0; i < 4; ++i) {
            int e = t + i * 256;              // 1024 float4 slots
            int kk = e >> 5, n4 = (e & 31) << 2;
            f32x4 v4 = *(const f32x4*)(W + (size_t)(k0 + kk) * D_MODEL + n0 + n4);
            Bs[(n4 + 0) * 40 + kk] = bf16b(v4[0]);
            Bs[(n4 + 1) * 40 + kk] = bf16b(v4[1]);
            Bs[(n4 + 2) * 40 + kk] = bf16b(v4[2]);
            Bs[(n4 + 3) * 40 + kk] = bf16b(v4[3]);
        }
        if (k0 + 32 < D_MODEL) {              // warm L2 for next K tile
            int r = t >> 3, c = (t & 7) << 2;
            __builtin_prefetch(X + (size_t)(m0 + r) * D_MODEL + k0 + 32 + c, 0, 1);
            __builtin_prefetch(W + (size_t)(k0 + 32 + (t >> 5)) * D_MODEL + n0 + ((t & 31) << 2), 0, 1);
        }
        __syncthreads();

        FragBF a[2], bf[4];
#pragma unroll
        for (int i = 0; i < 2; ++i) {
            int row = wm * 32 + i * 16 + ln;
            a[i].q[0] = *(const u32x4*)&As[row * 40 + 8 * g];
            a[i].q[1] = *(const u32x4*)&As[row * 40 + 16 + 8 * g];
        }
#pragma unroll
        for (int j = 0; j < 4; ++j) {
            int col = wn * 64 + j * 16 + ln;
            bf[j].q[0] = *(const u32x4*)&Bs[col * 40 + 8 * g];
            bf[j].q[1] = *(const u32x4*)&Bs[col * 40 + 16 + 8 * g];
        }
#pragma unroll
        for (int i = 0; i < 2; ++i)
#pragma unroll
            for (int j = 0; j < 4; ++j)
                acc[i][j] = __builtin_amdgcn_wmma_f32_16x16x32_bf16(
                    false, a[i].v, false, bf[j].v, (short)0, acc[i][j], false, false);
        __syncthreads();
    }

#pragma unroll
    for (int i = 0; i < 2; ++i)
#pragma unroll
        for (int j = 0; j < 4; ++j)
#pragma unroll
            for (int el = 0; el < 8; ++el) {
                int m = m0 + wm * 32 + i * 16 + g * 8 + el;
                int n = n0 + wn * 64 + j * 16 + ln;
                float val = acc[i][j][el] + bias[n];
                int bb = m >> 11, s = m & (SEQ - 1);
                int h = n >> 6, dk = n & 63;
                Out[(((size_t)(bb * NHEADS + h) * SEQ + s) * DKH) + dk] = bf16b(val);
            }
}

// ---------------------------------------------------------------------------
// Stage 2a: scores = (Qh @ Kh^T) * 0.125, causal-masked, fp32 -> att region
//           Q/K tiles staged via async global->LDS (pure bf16 copies)
// ---------------------------------------------------------------------------
__global__ __launch_bounds__(256)
void scores_kernel(const unsigned short* __restrict__ Qh, const unsigned short* __restrict__ Kh,
                   const int* __restrict__ mask, float* __restrict__ att)
{
    const int q0 = blockIdx.x * 128, k0 = blockIdx.y * 128, z = blockIdx.z;
    float* outp = att + (size_t)z * SEQ * SEQ;
    const int t = threadIdx.x;

    if (k0 > q0 + 127) {                       // tile fully above diagonal
        f32x4 nv = { NEGV, NEGV, NEGV, NEGV };
#pragma unroll
        for (int i = 0; i < 16; ++i) {
            int e = t + i * 256;               // 4096 f32x4 slots
            int r = e >> 5, c = (e & 31) << 2;
            *(f32x4*)(outp + (size_t)(q0 + r) * SEQ + k0 + c) = nv;
        }
        return;
    }

    __shared__ unsigned short As[128 * 72];    // [q][dk], stride 72 (144B, 16B aligned)
    __shared__ unsigned short Bs[128 * 72];    // [k][dk]
    const unsigned short* Qp = Qh + (size_t)z * SEQ * DKH;
    const unsigned short* Kp = Kh + (size_t)z * SEQ * DKH;
    const int lane = t & 31, w = t >> 5, g = lane >> 4, ln = lane & 15;
    const int wm = w >> 1, wn = w & 1;

#pragma unroll
    for (int i = 0; i < 8; ++i) {
        int e = t + i * 256;                   // 2048 8-byte slots (128 rows x 64 bf16)
        int r = e >> 4, c = (e & 15) << 2;
        async_ld_b64((unsigned)(uintptr_t)&As[r * 72 + c], Qp,
                     (unsigned)(((q0 + r) * DKH + c) * 2));
        async_ld_b64((unsigned)(uintptr_t)&Bs[r * 72 + c], Kp,
                     (unsigned)(((k0 + r) * DKH + c) * 2));
    }
    wait_async0();
    __syncthreads();

    v8f acc[2][4];
#pragma unroll
    for (int i = 0; i < 2; ++i)
#pragma unroll
        for (int j = 0; j < 4; ++j) zero8(acc[i][j]);

#pragma unroll
    for (int kk = 0; kk < 64; kk += 32) {
        FragBF a[2], bf[4];
#pragma unroll
        for (int i = 0; i < 2; ++i) {
            int row = wm * 32 + i * 16 + ln;
            a[i].q[0] = *(const u32x4*)&As[row * 72 + kk + 8 * g];
            a[i].q[1] = *(const u32x4*)&As[row * 72 + kk + 16 + 8 * g];
        }
#pragma unroll
        for (int j = 0; j < 4; ++j) {
            int col = wn * 64 + j * 16 + ln;
            bf[j].q[0] = *(const u32x4*)&Bs[col * 72 + kk + 8 * g];
            bf[j].q[1] = *(const u32x4*)&Bs[col * 72 + kk + 16 + 8 * g];
        }
#pragma unroll
        for (int i = 0; i < 2; ++i)
#pragma unroll
            for (int j = 0; j < 4; ++j)
                acc[i][j] = __builtin_amdgcn_wmma_f32_16x16x32_bf16(
                    false, a[i].v, false, bf[j].v, (short)0, acc[i][j], false, false);
    }

    const int b = z >> 4;
#pragma unroll
    for (int i = 0; i < 2; ++i)
#pragma unroll
        for (int j = 0; j < 4; ++j)
#pragma unroll
            for (int el = 0; el < 8; ++el) {
                int q = q0 + wm * 32 + i * 16 + g * 8 + el;
                int k = k0 + wn * 64 + j * 16 + ln;
                float val = acc[i][j][el] * 0.125f;
                int mv = mask[((size_t)b * SEQ + q) * SEQ + k];
                if (mv == 0) val = NEGV;
                outp[(size_t)q * SEQ + k] = val;
            }
}

// ---------------------------------------------------------------------------
// Stage 2b: row softmax, in place (one block per row)
// ---------------------------------------------------------------------------
__global__ __launch_bounds__(256)
void softmax_kernel(float* __restrict__ att)
{
    const int t = threadIdx.x;
    float* p = att + (size_t)blockIdx.x * SEQ;
    __shared__ float red[256];

    float loc[8]; float mx = -3.4e38f;
#pragma unroll
    for (int i = 0; i < 8; ++i) { loc[i] = p[t + i * 256]; mx = fmaxf(mx, loc[i]); }
    red[t] = mx; __syncthreads();
    for (int off = 128; off > 0; off >>= 1) {
        if (t < off) red[t] = fmaxf(red[t], red[t + off]);
        __syncthreads();
    }
    mx = red[0]; __syncthreads();

    float sum = 0.0f;
#pragma unroll
    for (int i = 0; i < 8; ++i) { loc[i] = __expf(loc[i] - mx); sum += loc[i]; }
    red[t] = sum; __syncthreads();
    for (int off = 128; off > 0; off >>= 1) {
        if (t < off) red[t] += red[t + off];
        __syncthreads();
    }
    float inv = 1.0f / red[0];
#pragma unroll
    for (int i = 0; i < 8; ++i) p[t + i * 256] = loc[i] * inv;
}

// ---------------------------------------------------------------------------
// Stage 3: ctx = attention @ Vh   (K loop truncated at causal boundary)
// ---------------------------------------------------------------------------
__global__ __launch_bounds__(256)
void attnv_kernel(const float* __restrict__ att, const unsigned short* __restrict__ Vh,
                  unsigned short* __restrict__ ctx)
{
    __shared__ unsigned short As[128 * 40];    // [q][k] bf16 probs
    __shared__ unsigned short Bs[64 * 40];     // [dk][k] bf16 V^T
    const int q0 = blockIdx.x * 128, z = blockIdx.z;
    const int b = z >> 4, h = z & 15;
    const float* P = att + (size_t)z * SEQ * SEQ;
    const unsigned short* Vp = Vh + (size_t)z * SEQ * DKH;
    const int t = threadIdx.x, lane = t & 31, w = t >> 5;
    const int g = lane >> 4, ln = lane & 15;
    const int wm = w >> 1, wn = w & 1;         // 4x2 waves -> 32x32 per wave

    v8f acc[2][2];
#pragma unroll
    for (int i = 0; i < 2; ++i)
#pragma unroll
        for (int j = 0; j < 2; ++j) zero8(acc[i][j]);

    const int kmax = q0 + 128;                 // probs are ~0 beyond the diagonal
    for (int k0 = 0; k0 < kmax; k0 += 32) {
#pragma unroll
        for (int i = 0; i < 4; ++i) {
            int e = t + i * 256;               // 1024 f32x4 slots (128x32 fp32)
            int r = e >> 3, c = (e & 7) << 2;
            f32x4 v4 = *(const f32x4*)(P + (size_t)(q0 + r) * SEQ + k0 + c);
            u32x2 pk; pk.x = pack2(v4[0], v4[1]); pk.y = pack2(v4[2], v4[3]);
            *(u32x2*)&As[r * 40 + c] = pk;
        }
#pragma unroll
        for (int i = 0; i < 8; ++i) {
            int e = t + i * 256;               // 2048 elems (32k x 64dk), transpose into Bs
            int kk = e >> 6, dk = e & 63;
            Bs[dk * 40 + kk] = Vp[(size_t)(k0 + kk) * DKH + dk];
        }
        __syncthreads();

        FragBF a[2], bf[2];
#pragma unroll
        for (int i = 0; i < 2; ++i) {
            int row = wm * 32 + i * 16 + ln;
            a[i].q[0] = *(const u32x4*)&As[row * 40 + 8 * g];
            a[i].q[1] = *(const u32x4*)&As[row * 40 + 16 + 8 * g];
        }
#pragma unroll
        for (int j = 0; j < 2; ++j) {
            int col = wn * 32 + j * 16 + ln;
            bf[j].q[0] = *(const u32x4*)&Bs[col * 40 + 8 * g];
            bf[j].q[1] = *(const u32x4*)&Bs[col * 40 + 16 + 8 * g];
        }
#pragma unroll
        for (int i = 0; i < 2; ++i)
#pragma unroll
            for (int j = 0; j < 2; ++j)
                acc[i][j] = __builtin_amdgcn_wmma_f32_16x16x32_bf16(
                    false, a[i].v, false, bf[j].v, (short)0, acc[i][j], false, false);
        __syncthreads();
    }

#pragma unroll
    for (int i = 0; i < 2; ++i)
#pragma unroll
        for (int j = 0; j < 2; ++j)
#pragma unroll
            for (int el = 0; el < 8; ++el) {
                int q = q0 + wm * 32 + i * 16 + g * 8 + el;
                int n = wn * 32 + j * 16 + ln;
                ctx[((size_t)(b * SEQ + q)) * D_MODEL + h * DKH + n] = bf16b(acc[i][j][el]);
            }
}

// ---------------------------------------------------------------------------
// Stage 4: x = ctx @ Wo + bo  (fp32 out); ctx tile staged via async LDS copy
// ---------------------------------------------------------------------------
__global__ __launch_bounds__(256)
void oproj_kernel(const unsigned short* __restrict__ ctx, const float* __restrict__ W,
                  const float* __restrict__ bias, float* __restrict__ out)
{
    __shared__ unsigned short As[128 * 40];
    __shared__ unsigned short Bs[128 * 40];
    const int m0 = blockIdx.x * 128, n0 = blockIdx.y * 128;
    const int t = threadIdx.x, lane = t & 31, w = t >> 5;
    const int g = lane >> 4, ln = lane & 15;
    const int wm = w >> 1, wn = w & 1;

    v8f acc[2][4];
#pragma unroll
    for (int i = 0; i < 2; ++i)
#pragma unroll
        for (int j = 0; j < 4; ++j) zero8(acc[i][j]);

    for (int k0 = 0; k0 < D_MODEL; k0 += 32) {
        // A tile: pure bf16 copy -> async global->LDS
#pragma unroll
        for (int i = 0; i < 4; ++i) {
            int e = t + i * 256;               // 1024 8-byte slots (128x32 bf16)
            int r = e >> 3, c = (e & 7) << 2;
            async_ld_b64((unsigned)(uintptr_t)&As[r * 40 + c], ctx,
                         (unsigned)((((m0 + r) * D_MODEL) + k0 + c) * 2));
        }
        // B tile: fp32 -> bf16 conversion path
#pragma unroll
        for (int i = 0; i < 4; ++i) {
            int e = t + i * 256;
            int kk = e >> 5, n4 = (e & 31) << 2;
            f32x4 v4 = *(const f32x4*)(W + (size_t)(k0 + kk) * D_MODEL + n0 + n4);
            Bs[(n4 + 0) * 40 + kk] = bf16b(v4[0]);
            Bs[(n4 + 1) * 40 + kk] = bf16b(v4[1]);
            Bs[(n4 + 2) * 40 + kk] = bf16b(v4[2]);
            Bs[(n4 + 3) * 40 + kk] = bf16b(v4[3]);
        }
        if (k0 + 32 < D_MODEL) {
            __builtin_prefetch(W + (size_t)(k0 + 32 + (t >> 5)) * D_MODEL + n0 + ((t & 31) << 2), 0, 1);
        }
        wait_async0();
        __syncthreads();

        FragBF a[2], bf[4];
#pragma unroll
        for (int i = 0; i < 2; ++i) {
            int row = wm * 32 + i * 16 + ln;
            a[i].q[0] = *(const u32x4*)&As[row * 40 + 8 * g];
            a[i].q[1] = *(const u32x4*)&As[row * 40 + 16 + 8 * g];
        }
#pragma unroll
        for (int j = 0; j < 4; ++j) {
            int col = wn * 64 + j * 16 + ln;
            bf[j].q[0] = *(const u32x4*)&Bs[col * 40 + 8 * g];
            bf[j].q[1] = *(const u32x4*)&Bs[col * 40 + 16 + 8 * g];
        }
#pragma unroll
        for (int i = 0; i < 2; ++i)
#pragma unroll
            for (int j = 0; j < 4; ++j)
                acc[i][j] = __builtin_amdgcn_wmma_f32_16x16x32_bf16(
                    false, a[i].v, false, bf[j].v, (short)0, acc[i][j], false, false);
        __syncthreads();
    }

#pragma unroll
    for (int i = 0; i < 2; ++i)
#pragma unroll
        for (int j = 0; j < 4; ++j)
#pragma unroll
            for (int el = 0; el < 8; ++el) {
                int m = m0 + wm * 32 + i * 16 + g * 8 + el;
                int n = n0 + wn * 64 + j * 16 + ln;
                out[(size_t)m * D_MODEL + n] = acc[i][j][el] + bias[n];
            }
}

// ---------------------------------------------------------------------------
extern "C" void kernel_launch(void* const* d_in, const int* in_sizes, int n_in,
                              void* d_out, int out_size, void* d_ws, size_t ws_size,
                              hipStream_t stream)
{
    const float* query = (const float*)d_in[0];
    const float* key_  = (const float*)d_in[1];
    const float* value = (const float*)d_in[2];
    const int*   mask  = (const int*)d_in[3];
    const float* Wq = (const float*)d_in[4];
    const float* bq = (const float*)d_in[5];
    const float* Wk = (const float*)d_in[6];
    const float* bk = (const float*)d_in[7];
    const float* Wv = (const float*)d_in[8];
    const float* bv = (const float*)d_in[9];
    const float* Wo = (const float*)d_in[10];
    const float* bo = (const float*)d_in[11];

    float* out_x   = (float*)d_out;
    float* out_att = out_x + (size_t)ROWS * D_MODEL;

    unsigned short* ws  = (unsigned short*)d_ws;
    const size_t HS = (size_t)NB * NHEADS * SEQ * DKH;   // 4,194,304 elems (8MB bf16)
    unsigned short* Qh  = ws;
    unsigned short* Kh  = ws + HS;
    unsigned short* Vh  = ws + 2 * HS;
    unsigned short* Ctx = ws + 3 * HS;

    dim3 blk(256);
    qkv_proj_kernel<<<dim3(32, 8, 1), blk, 0, stream>>>(query, Wq, bq, Qh);
    qkv_proj_kernel<<<dim3(32, 8, 1), blk, 0, stream>>>(key_,  Wk, bk, Kh);
    qkv_proj_kernel<<<dim3(32, 8, 1), blk, 0, stream>>>(value, Wv, bv, Vh);
    scores_kernel<<<dim3(16, 16, 32), blk, 0, stream>>>(Qh, Kh, mask, out_att);
    softmax_kernel<<<dim3(NB * NHEADS * SEQ, 1, 1), blk, 0, stream>>>(out_att);
    attnv_kernel<<<dim3(16, 1, 32), blk, 0, stream>>>(out_att, Vh, Ctx);
    oproj_kernel<<<dim3(32, 8, 1), blk, 0, stream>>>(Ctx, Wo, bo, out_x);
}
